// GatedSpatialCrossAttention_75084618269430
// MI455X (gfx1250) — matrix operand order repaired and checked by reading, not compiled, not run on hardware
//
#include <hip/hip_runtime.h>
#include <math.h>

// ---------------------------------------------------------------------------
// GatedSpatialCrossAttention for MI455X (gfx1250, wave32, WMMA)
// B=32 L=256 S=64 D=128 H=4 DH=32
//
// Dominant cost: K/V projection of h_all (34.4 GFLOP, 268 MB read) -> done in
// bf16 with v_wmma_f32_16x16x32_bf16, f32 accumulate. Everything else is
// small and done in fp32 VALU, fully fused per (b,l) row so h_all is read
// exactly once from HBM (memory-bound floor ~11.5 us at 23.3 TB/s).
// ---------------------------------------------------------------------------

typedef __attribute__((ext_vector_type(16))) __bf16 v16bf;
typedef __attribute__((ext_vector_type(8)))  float  v8f;
typedef __attribute__((ext_vector_type(4)))  unsigned int u32x4;

union Frag16 { v16bf v; u32x4 u[2]; };

static __device__ __forceinline__ __bf16 f2bf(float f) {
  unsigned int u = __float_as_uint(f);
  u += 0x7FFFu + ((u >> 16) & 1u);          // round-to-nearest-even
  unsigned short h = (unsigned short)(u >> 16);
  __bf16 r;
  __builtin_memcpy(&r, &h, 2);
  return r;
}

#define Bc   32
#define Lc   256
#define Sc   64
#define Dc   128
#define Hc   4
#define DHc  32

#define SAP  136   // bf16 pitch for A tile   (64  x 128)
#define SWP  136   // bf16 pitch for weights  (128 x 128)
#define SKP  132   // f32 pitch for K/V       (64  x 128)

// ---------------------------------------------------------------------------
// Kernel 1: convert kw / vw (rows 128..383 of in_proj_w) to bf16 in workspace
// ---------------------------------------------------------------------------
__global__ void convert_weights_kernel(const float* __restrict__ in_proj_w,
                                       __bf16* __restrict__ kwbf,
                                       __bf16* __restrict__ vwbf) {
  int t = blockIdx.x * blockDim.x + threadIdx.x;
  if (t < Dc * Dc) {
    kwbf[t] = f2bf(in_proj_w[Dc * Dc + t]);      // kw rows 128..255
    vwbf[t] = f2bf(in_proj_w[2 * Dc * Dc + t]);  // vw rows 256..383
  }
}

// ---------------------------------------------------------------------------
// Kernel 2: adjacency mask bias per (b, s): 0 if allowed, -inf otherwise
// ---------------------------------------------------------------------------
__global__ void mask_kernel(const int* __restrict__ station_id,
                            const float* __restrict__ adj,
                            float* __restrict__ maskbias) {
  __shared__ int cnt1, cnt2;
  int b = blockIdx.x;
  int s = threadIdx.x;  // 0..63
  if (s == 0) { cnt1 = 0; cnt2 = 0; }
  __syncthreads();
  int sid = station_id[b];
  bool a0 = (adj[sid * Sc + s] > 0.0f) && (s != sid);
  if (a0) atomicAdd(&cnt1, 1);
  __syncthreads();
  bool allowed = (cnt1 > 0) ? a0 : (s != sid);       // fallback-to-all, no self
  if (allowed) atomicAdd(&cnt2, 1);
  __syncthreads();
  if (cnt2 == 0) allowed = allowed || (s == sid);    // S==1 safety
  maskbias[b * Sc + s] = allowed ? 0.0f : -INFINITY;
}

// ---------------------------------------------------------------------------
// Kernel 3: fused attention, one workgroup (8 wave32) per (b,l) row
// ---------------------------------------------------------------------------
__global__ __launch_bounds__(256)
void fused_attn_kernel(const float* __restrict__ h_wq,
                       const float* __restrict__ h_all,
                       const float* __restrict__ in_b,
                       const float* __restrict__ in_proj_w,   // qw = rows 0..127 (f32)
                       const float* __restrict__ out_w,
                       const float* __restrict__ out_b,
                       const float* __restrict__ gate_w,
                       const float* __restrict__ gate_b,
                       const float* __restrict__ ln_g,
                       const float* __restrict__ ln_b,
                       const float* __restrict__ alpha_p,
                       const float* __restrict__ maskbias,
                       const __bf16* __restrict__ kwbf,
                       const __bf16* __restrict__ vwbf,
                       float* __restrict__ out) {
  __shared__ __align__(16) __bf16 sA[Sc * SAP];   // h_all tile, bf16
  __shared__ __align__(16) __bf16 sW[Dc * SWP];   // current weight (kw then vw)
  __shared__ __align__(16) float  sK[Sc * SKP];
  __shared__ __align__(16) float  sV[Sc * SKP];
  __shared__ float sHwq[Dc], sQ[Dc];
  __shared__ float sScores[Hc * Sc], sWgt[Hc * Sc], sM[Hc], sSum[Hc];
  __shared__ float sCtx[Dc], sCtxO[Dc], sY[Dc];
  __shared__ float sRed1[Dc], sRed2[Dc];

  const int t    = threadIdx.x;
  const int wave = t >> 5;
  const int lane = t & 31;
  const int row  = blockIdx.x;        // b*L + l
  const int b    = row >> 8;          // L == 256

  // ---- Phase 0: loads -----------------------------------------------------
  if (t < Dc) sHwq[t] = h_wq[(size_t)row * Dc + t];

  {
    const float4* src = (const float4*)(h_all + (size_t)row * (Sc * Dc));
#pragma unroll
    for (int it = 0; it < 8; ++it) {
      int f4 = t + it * 256;          // 0..2047
      int s  = f4 >> 5;               // 32 float4 per station row
      int c4 = (f4 & 31) * 4;
      float4 v = src[f4];
      __bf16* d = &sA[s * SAP + c4];
      d[0] = f2bf(v.x); d[1] = f2bf(v.y); d[2] = f2bf(v.z); d[3] = f2bf(v.w);
    }
  }
  {
    const u32x4* s4 = (const u32x4*)kwbf;   // 2048 x (8 bf16)
#pragma unroll
    for (int it = 0; it < 8; ++it) {
      int f = t + it * 256;
      int j = f >> 4;
      int c8 = (f & 15) * 8;
      *(u32x4*)&sW[j * SWP + c8] = s4[f];
    }
  }
  __syncthreads();

  // ---- Phase 1: Q = h_wq @ qw.T + qb  (fp32 VALU, small) ------------------
  if (t < Dc) {
    float acc = in_b[t];
    const float* wr = &in_proj_w[t * Dc];
    for (int d2 = 0; d2 < Dc; ++d2) acc += sHwq[d2] * wr[d2];
    sQ[t] = acc;
  }

  // ---- Phase 2: K/V projection via WMMA bf16 ------------------------------
  // wave -> (station tile st = wave&3, n-tile base ntb = (wave>>2)*4)
  const int st  = wave & 3;
  const int ntb = (wave >> 2) * 4;

  // Preload 4 A fragments (one per K-step); shared by K and V projections.
  // 16-bit A 16x32 layout: lane<16 row m holds K {0..7,16..23}; lane>=16 {8..15,24..31}
  Frag16 afrag[4];
  {
    const int mrow  = lane & 15;
    const int kbase = (lane >> 4) << 3;   // 0 or 8
    const __bf16* rowp = &sA[(st * 16 + mrow) * SAP + kbase];
#pragma unroll
    for (int ks = 0; ks < 4; ++ks) {
      const __bf16* p = rowp + ks * 32;
      afrag[ks].u[0] = *(const u32x4*)(p);
      afrag[ks].u[1] = *(const u32x4*)(p + 16);
    }
  }

  const int ncol = lane & 15;
  const int koff = (lane >> 4) << 3;      // B: lane<16 holds K {0..7,16..23}, else {8..15,24..31}
  const int mb   = (lane >> 4) << 3;      // D: lane<16 rows 0..7, else 8..15

  // K projection
#pragma unroll
  for (int nti = 0; nti < 4; ++nti) {
    int nt = ntb + nti;
    v8f acc = {0.f, 0.f, 0.f, 0.f, 0.f, 0.f, 0.f, 0.f};
#pragma unroll
    for (int ks = 0; ks < 4; ++ks) {
      Frag16 bf;
      const __bf16* p = &sW[(nt * 16 + ncol) * SWP + ks * 32 + koff];
      bf.u[0] = *(const u32x4*)(p);
      bf.u[1] = *(const u32x4*)(p + 16);
      acc = __builtin_amdgcn_wmma_f32_16x16x32_bf16(
          false, afrag[ks].v, false, bf.v, (short)0, acc, false, false);
    }
#pragma unroll
    for (int r = 0; r < 8; ++r)
      sK[(st * 16 + mb + r) * SKP + nt * 16 + ncol] = acc[r];
  }
  __syncthreads();

  // swap weights to vw
  {
    const u32x4* s4 = (const u32x4*)vwbf;
#pragma unroll
    for (int it = 0; it < 8; ++it) {
      int f = t + it * 256;
      int j = f >> 4;
      int c8 = (f & 15) * 8;
      *(u32x4*)&sW[j * SWP + c8] = s4[f];
    }
  }
  __syncthreads();

  // V projection (reuses A fragments held in VGPRs)
#pragma unroll
  for (int nti = 0; nti < 4; ++nti) {
    int nt = ntb + nti;
    v8f acc = {0.f, 0.f, 0.f, 0.f, 0.f, 0.f, 0.f, 0.f};
#pragma unroll
    for (int ks = 0; ks < 4; ++ks) {
      Frag16 bf;
      const __bf16* p = &sW[(nt * 16 + ncol) * SWP + ks * 32 + koff];
      bf.u[0] = *(const u32x4*)(p);
      bf.u[1] = *(const u32x4*)(p + 16);
      acc = __builtin_amdgcn_wmma_f32_16x16x32_bf16(
          false, afrag[ks].v, false, bf.v, (short)0, acc, false, false);
    }
#pragma unroll
    for (int r = 0; r < 8; ++r)
      sV[(st * 16 + mb + r) * SKP + nt * 16 + ncol] = acc[r];
  }
  __syncthreads();

  // ---- Phase 3: add K/V biases -------------------------------------------
  {
    const float* kb = in_b + Dc;
    const float* vb = in_b + 2 * Dc;
    for (int idx = t; idx < Sc * Dc; idx += 256) {
      int s = idx >> 7, j = idx & 127;
      sK[s * SKP + j] += kb[j];
      sV[s * SKP + j] += vb[j];
    }
  }
  __syncthreads();

  // ---- Phase 4: scores + masked softmax -----------------------------------
  {
    int h = t >> 6, s = t & 63;
    float acc = 0.f;
    const float* qv = &sQ[h * DHc];
    const float* kv = &sK[s * SKP + h * DHc];
#pragma unroll
    for (int i = 0; i < DHc; ++i) acc += qv[i] * kv[i];
    sScores[h * Sc + s] = acc * 0.1767766952966369f /* 1/sqrt(32) */
                        + maskbias[b * Sc + s];
  }
  __syncthreads();
  if (t < Hc) {
    float m = -INFINITY;
    for (int s = 0; s < Sc; ++s) m = fmaxf(m, sScores[t * Sc + s]);
    float sum = 0.f;
    for (int s = 0; s < Sc; ++s) sum += __expf(sScores[t * Sc + s] - m);
    sM[t] = m; sSum[t] = sum;
  }
  __syncthreads();
  {
    int h = t >> 6, s = t & 63;
    sWgt[h * Sc + s] = __expf(sScores[h * Sc + s] - sM[h]) / sSum[h];
  }
  __syncthreads();

  // ---- Phase 5: ctx = w @ V ----------------------------------------------
  if (t < Dc) {
    int h = t >> 5;
    float acc = 0.f;
    for (int s = 0; s < Sc; ++s) acc += sWgt[h * Sc + s] * sV[s * SKP + t];
    sCtx[t] = acc;
  }
  __syncthreads();

  // ---- Phase 6: out projection -------------------------------------------
  if (t < Dc) {
    float acc = out_b[t];
    const float* wr = &out_w[t * Dc];
    for (int d2 = 0; d2 < Dc; ++d2) acc += sCtx[d2] * wr[d2];
    sCtxO[t] = acc;
  }
  __syncthreads();

  // ---- Phase 7: gate + residual ------------------------------------------
  const float alpha = alpha_p[0];
  if (t < Dc) {
    float acc = gate_b[t];
    const float* wr = &gate_w[t * 2 * Dc];
    for (int d2 = 0; d2 < Dc; ++d2) acc += sHwq[d2] * wr[d2];
    for (int d2 = 0; d2 < Dc; ++d2) acc += sCtxO[d2] * wr[Dc + d2];
    float g = 1.f / (1.f + __expf(-acc));
    float y = sHwq[t] + alpha * g * sCtxO[t];
    sY[t] = y; sRed1[t] = y; sRed2[t] = y * y;
  }
  __syncthreads();

  // ---- Phase 8: LayerNorm -------------------------------------------------
  for (int off = 64; off > 0; off >>= 1) {
    if (t < off) { sRed1[t] += sRed1[t + off]; sRed2[t] += sRed2[t + off]; }
    __syncthreads();
  }
  if (t < Dc) {
    float mu  = sRed1[0] * (1.f / Dc);
    float var = sRed2[0] * (1.f / Dc) - mu * mu;
    float ny  = (sY[t] - mu) * rsqrtf(var + 1e-5f);
    out[(size_t)row * Dc + t] = ny * ln_g[t] + ln_b[t];
  }
}

// ---------------------------------------------------------------------------
extern "C" void kernel_launch(void* const* d_in, const int* in_sizes, int n_in,
                              void* d_out, int out_size, void* d_ws, size_t ws_size,
                              hipStream_t stream) {
  const float* h_wq   = (const float*)d_in[0];
  const float* h_all  = (const float*)d_in[1];
  const int*   sid    = (const int*)d_in[2];
  const float* adj    = (const float*)d_in[3];
  const float* ipw    = (const float*)d_in[4];
  const float* ipb    = (const float*)d_in[5];
  const float* out_w  = (const float*)d_in[6];
  const float* out_b  = (const float*)d_in[7];
  const float* gate_w = (const float*)d_in[8];
  const float* gate_b = (const float*)d_in[9];
  const float* ln_g   = (const float*)d_in[10];
  const float* ln_b   = (const float*)d_in[11];
  const float* alpha  = (const float*)d_in[12];
  float* out = (float*)d_out;

  float*  maskbias = (float*)d_ws;                                   // 8 KB
  __bf16* kwbf     = (__bf16*)((char*)d_ws + 8192);                  // 32 KB
  __bf16* vwbf     = (__bf16*)((char*)d_ws + 8192 + Dc * Dc * 2);    // 32 KB

  convert_weights_kernel<<<(Dc * Dc + 255) / 256, 256, 0, stream>>>(ipw, kwbf, vwbf);
  mask_kernel<<<Bc, Sc, 0, stream>>>(sid, adj, maskbias);
  fused_attn_kernel<<<Bc * Lc, 256, 0, stream>>>(h_wq, h_all, ipb, ipw,
                                                 out_w, out_b, gate_w, gate_b,
                                                 ln_g, ln_b, alpha, maskbias,
                                                 kwbf, vwbf, out);
}
